// GroupQueryAttention_62998580298173
// MI455X (gfx1250) — matrix operand order repaired and checked by reading, not compiled, not run on hardware
//
#include <hip/hip_runtime.h>

// ---------------------------------------------------------------------------
// GQA forward for MI455X (gfx1250, wave32, WMMA + Tensor Data Mover).
// Pipeline: f32->bf16 convert, TDM-fed WMMA GEMM projections, WMMA scores,
// row softmax (attn materialized f32 in d_out), WMMA attn*V, WMMA out-proj.
// ---------------------------------------------------------------------------

typedef __bf16 bf16;
typedef __attribute__((ext_vector_type(16))) __bf16        v16bf;
typedef __attribute__((ext_vector_type(8)))  float         v8f;
typedef __attribute__((ext_vector_type(4)))  unsigned int  u32x4;
typedef __attribute__((ext_vector_type(8)))  int           i32x8;
typedef __attribute__((ext_vector_type(4)))  int           i32x4;

#define B_  2
#define S_  2048
#define D_  1024
#define H_  16
#define DK_ 64

#if defined(__has_builtin)
#if __has_builtin(__builtin_amdgcn_tensor_load_to_lds) && \
    __has_builtin(__builtin_amdgcn_s_wait_tensorcnt)
#define HAS_TDM 1
#endif
#endif
#ifndef HAS_TDM
#define HAS_TDM 0
#endif

#if HAS_TDM
// ---------------------------------------------------------------------------
// Issue a TDM 2D tile load (global -> LDS) per the CDNA5 D# layout.
// Tile rows are tile_d0 bf16 elements at stride stride0_elems; LDS rows get
// DMA-inserted padding: after every (2<<pad_icode) DWORDs add (pad_acode+1)
// DWORDs, matching our padded LDS tile arrays. data_size = 2 bytes.
// This toolchain's builtin is the 6-arg form:
//   (u32x4 g0, i32x8 g1, i32x4 g2, i32x4 g3, i32x8, i32 cpol)
// ---------------------------------------------------------------------------
__device__ __forceinline__ void tdm_load_tile_2d(void* lds_ptr, const bf16* gptr,
                                                 unsigned tile_d0, unsigned tile_d1,
                                                 unsigned stride0_elems,
                                                 unsigned pad_icode, unsigned pad_acode)
{
    const unsigned long long ga = (unsigned long long)gptr;
    const unsigned lds_addr = (unsigned)(unsigned long long)lds_ptr; // low 32b = LDS byte addr

    u32x4 g0;
    g0.x = 1u;                                                   // count=1 (valid user D#)
    g0.y = lds_addr;                                             // LDS dest
    g0.z = (unsigned)ga;                                         // global_addr[31:0]
    g0.w = (unsigned)((ga >> 32) & 0x01FFFFFFu) | (2u << 30);    // addr[56:32] | type=2

    i32x8 g1;
    g1[0] = (int)((1u << 16)            // data_size = 2 bytes
                | (1u << 20)            // pad_enable
                | (pad_icode << 22)     // pad_interval
                | (pad_acode << 25));   // pad_amount
    g1[1] = (int)((tile_d0 & 0xFFFFu) << 16);                        // tensor_dim0 lo
    g1[2] = (int)(((tile_d0 >> 16) & 0xFFFFu) | ((tile_d1 & 0xFFFFu) << 16)); // dim0 hi | dim1 lo
    g1[3] = (int)(((tile_d1 >> 16) & 0xFFFFu) | ((tile_d0 & 0xFFFFu) << 16)); // dim1 hi | tile_dim0
    g1[4] = (int)(tile_d1 & 0xFFFFu);                                // tile_dim1 | tile_dim2=0 (2D)
    g1[5] = (int)stride0_elems;                                      // tensor_dim0_stride lo32
    g1[6] = 0;                                                       // stride hi | dim1_stride
    g1[7] = 0;

    const i32x4 z4 = {0, 0, 0, 0};                                   // groups 2/3 unused (2D)
    const i32x8 z8 = {0, 0, 0, 0, 0, 0, 0, 0};
    __builtin_amdgcn_tensor_load_to_lds(g0, g1, z4, z4, z8, 0);
}
#endif

// ---------------------------------------------------------------------------
__global__ __launch_bounds__(256)
void cvt_f32_bf16(const float* __restrict__ src, bf16* __restrict__ dst, int n) {
    int i = blockIdx.x * 256 + threadIdx.x;
    if (i < n) dst[i] = (bf16)src[i];
}

// ---------------------------------------------------------------------------
// out[m,n] = scale * sum_k A[m,k] * W[n,k]  (+ bias[n])
// A: [M,K] bf16 (stride lda); W: [N,K] bf16 (stride ldw).
// 128x128 block tile, 8 waves, each wave 32(M) x 64(N) = 8 WMMA acc tiles.
// Tiles DMA'd by the TDM into double-buffered LDS; DMA for step k+1 overlaps
// the 8 WMMAs of step k. headmode: grid.z = (b,h), outf -> attn page.
// ---------------------------------------------------------------------------
__global__ __launch_bounds__(256)
void gemm_xwT_bf16(const bf16* __restrict__ A, int lda,
                   const bf16* __restrict__ W, int ldw,
                   const float* __restrict__ bias, float scale,
                   bf16* __restrict__ outb, float* __restrict__ outf, int ldo,
                   int M, int N, int K, int headmode)
{
    __shared__ bf16 ldsA[2][128][40];   // 32 cols + 16B pad (TDM pad: 16dw -> +4dw)
    __shared__ bf16 ldsW[2][128][40];

    if (headmode) {
        const int z = blockIdx.z, b = z >> 4, h = z & 15;
        A    += (long)b * S_ * D_ + h * DK_;
        W    += (long)b * S_ * D_ + h * DK_;
        outf += (long)z * S_ * (long)S_;
    }

    const int tid  = threadIdx.x;
    const int lane = tid & 31;
    const int wave = tid >> 5;
    const int wr   = wave >> 1;            // 0..3 : M offset wr*32
    const int wc   = wave & 1;             // 0..1 : N offset wc*64
    const int m0   = blockIdx.y * 128;
    const int n0   = blockIdx.x * 128;

    const int arow0 = wr * 32 + (lane & 15);
    const int akb   = (lane >> 4) * 16;    // A K-groups 0..15 / 16..31
    const int bcol  = wc * 64 + (lane & 15);
    const int bkb   = (lane >> 4) * 16;    // B K-groups 0..15 / 16..31

    v8f acc[2][4] = {{{}, {}, {}, {}}, {{}, {}, {}, {}}};

#if HAS_TDM
    if (wave == 0) {                        // preload buffer 0
        tdm_load_tile_2d(&ldsA[0][0][0], &A[(long)m0 * lda], 32, 128, (unsigned)lda, 3, 3);
        tdm_load_tile_2d(&ldsW[0][0][0], &W[(long)n0 * ldw], 32, 128, (unsigned)ldw, 3, 3);
        __builtin_amdgcn_s_wait_tensorcnt(0);
    }
    __syncthreads();
#else
    {
        const int ldr = tid >> 2, ldc = (tid & 3) * 8;
#pragma unroll
        for (int r = 0; r < 2; ++r) {
            *(uint4*)&ldsA[0][ldr + r * 64][ldc] = *(const uint4*)&A[(long)(m0 + ldr + r * 64) * lda + ldc];
            *(uint4*)&ldsW[0][ldr + r * 64][ldc] = *(const uint4*)&W[(long)(n0 + ldr + r * 64) * ldw + ldc];
        }
    }
    __syncthreads();
#endif

    for (int kk = 0; kk < K; kk += 32) {
        const int  cur  = (kk >> 5) & 1, nxt = cur ^ 1;
        const bool more = (kk + 32) < K;

#if HAS_TDM
        if (wave == 0 && more) {            // overlap next-tile DMA with compute
            tdm_load_tile_2d(&ldsA[nxt][0][0], &A[(long)m0 * lda + kk + 32], 32, 128, (unsigned)lda, 3, 3);
            tdm_load_tile_2d(&ldsW[nxt][0][0], &W[(long)n0 * ldw + kk + 32], 32, 128, (unsigned)ldw, 3, 3);
        }
#else
        if (more) {
            const int ldr = tid >> 2, ldc = (tid & 3) * 8;
#pragma unroll
            for (int r = 0; r < 2; ++r) {
                *(uint4*)&ldsA[nxt][ldr + r * 64][ldc] = *(const uint4*)&A[(long)(m0 + ldr + r * 64) * lda + kk + 32 + ldc];
                *(uint4*)&ldsW[nxt][ldr + r * 64][ldc] = *(const uint4*)&W[(long)(n0 + ldr + r * 64) * ldw + kk + 32 + ldc];
            }
        }
#endif

        v16bf af[2];
#pragma unroll
        for (int mt = 0; mt < 2; ++mt)
#pragma unroll
            for (int j = 0; j < 8; ++j) {
                af[mt][j]     = ldsA[cur][arow0 + mt * 16][akb + j];
                af[mt][8 + j] = ldsA[cur][arow0 + mt * 16][akb + 8 + j];
            }
#pragma unroll
        for (int nt = 0; nt < 4; ++nt) {
            v16bf bfr;
#pragma unroll
            for (int j = 0; j < 16; ++j) bfr[j] = ldsW[cur][bcol + nt * 16][bkb + j];
            acc[0][nt] = __builtin_amdgcn_wmma_f32_16x16x32_bf16(false, af[0], false, bfr,
                                                                 (short)0, acc[0][nt], false, false);
            acc[1][nt] = __builtin_amdgcn_wmma_f32_16x16x32_bf16(false, af[1], false, bfr,
                                                                 (short)0, acc[1][nt], false, false);
        }
        __syncthreads();                    // all waves done reading buf[cur]
#if HAS_TDM
        if (wave == 0 && more) __builtin_amdgcn_s_wait_tensorcnt(0);
#endif
        __syncthreads();                    // buf[nxt] ready for everyone
    }

    // C/D layout: VGPR e -> M = e (lanes 0-15) / 8+e (lanes 16-31), N = lane&15
#pragma unroll
    for (int mt = 0; mt < 2; ++mt) {
        const int mrow = m0 + wr * 32 + mt * 16 + ((lane >> 4) << 3);
#pragma unroll
        for (int nt = 0; nt < 4; ++nt) {
            const int   ncol = n0 + wc * 64 + nt * 16 + (lane & 15);
            const float bb   = bias ? bias[ncol] : 0.f;
#pragma unroll
            for (int e = 0; e < 8; ++e) {
                const float v = acc[mt][nt][e] * scale + bb;
                if (outf) outf[(long)(mrow + e) * ldo + ncol] = v;
                else      outb[(long)(mrow + e) * ldo + ncol] = (bf16)v;
            }
        }
    }
}

// ---------------------------------------------------------------------------
// In-place row softmax over 2048-wide rows; one block per row.
// ---------------------------------------------------------------------------
__global__ __launch_bounds__(256)
void softmax_rows(float* __restrict__ attn)
{
    float* p = attn + (long)blockIdx.x * S_;
    const int tid = threadIdx.x;
    __shared__ float red[256];

    float vals[8];
    float m = -3.4e38f;
#pragma unroll
    for (int i = 0; i < 8; ++i) { vals[i] = p[tid + i * 256]; m = fmaxf(m, vals[i]); }
    red[tid] = m; __syncthreads();
    for (int s = 128; s > 0; s >>= 1) {
        if (tid < s) red[tid] = fmaxf(red[tid], red[tid + s]);
        __syncthreads();
    }
    m = red[0]; __syncthreads();

    float sum = 0.f;
#pragma unroll
    for (int i = 0; i < 8; ++i) { vals[i] = __expf(vals[i] - m); sum += vals[i]; }
    red[tid] = sum; __syncthreads();
    for (int s = 128; s > 0; s >>= 1) {
        if (tid < s) red[tid] += red[tid + s];
        __syncthreads();
    }
    const float inv = 1.f / red[0];
#pragma unroll
    for (int i = 0; i < 8; ++i) p[tid + i * 256] = vals[i] * inv;
}

// ---------------------------------------------------------------------------
// Per (b,h): AOut[q,d] = sum_k attn[q,k] * Vh[k,d].  M=2048, N=64, K=2048.
// Block = 128 query rows (8 waves x 16). V streamed in 128-key chunks by the
// TDM (double-buffered); attn converted f32->bf16 in-register per A fragment.
// 16 WMMAs per wave between barrier pairs.
// ---------------------------------------------------------------------------
__global__ __launch_bounds__(256)
void attn_pv(const float* __restrict__ attn, const bf16* __restrict__ Vh,
             bf16* __restrict__ AOut)
{
    __shared__ bf16 ldsV[2][128][72];       // [k][d] chunk, +16B row pad (TDM: 32dw -> +4dw)

    const int z = blockIdx.y, b = z >> 4, h = z & 15;
    const int tid = threadIdx.x, lane = tid & 31, wave = tid >> 5;
    const int mtile = blockIdx.x * 128 + wave * 16;

    const float* Arow = attn + ((long)z * S_ + (mtile + (lane & 15))) * S_;
    const bf16*  Vbase = Vh + (long)b * S_ * D_ + h * DK_;
    const int akb = (lane >> 4) * 16;
    const int col = lane & 15;
    const int bkb = (lane >> 4) * 16;

    v8f acc[4] = {{}, {}, {}, {}};

#if HAS_TDM
    if (wave == 0) {
        tdm_load_tile_2d(&ldsV[0][0][0], Vbase, 64, 128, D_, 4, 3);
        __builtin_amdgcn_s_wait_tensorcnt(0);
    }
    __syncthreads();
#else
    {
        const int vr = tid >> 1, vc = (tid & 1) * 32;
#pragma unroll
        for (int c = 0; c < 4; ++c)
            *(uint4*)&ldsV[0][vr][vc + c * 8] = *(const uint4*)&Vbase[(long)vr * D_ + vc + c * 8];
    }
    __syncthreads();
#endif

    for (int kc = 0; kc < S_; kc += 128) {
        const int  cur  = (kc >> 7) & 1, nxt = cur ^ 1;
        const bool more = (kc + 128) < S_;

#if HAS_TDM
        if (wave == 0 && more)
            tdm_load_tile_2d(&ldsV[nxt][0][0], Vbase + (long)(kc + 128) * D_, 64, 128, D_, 4, 3);
#else
        if (more) {
            const int vr = tid >> 1, vc = (tid & 1) * 32;
#pragma unroll
            for (int c = 0; c < 4; ++c)
                *(uint4*)&ldsV[nxt][vr][vc + c * 8] =
                    *(const uint4*)&Vbase[(long)(kc + 128 + vr) * D_ + vc + c * 8];
        }
#endif

#pragma unroll
        for (int ks = 0; ks < 4; ++ks) {    // 4 x K=32 steps inside the chunk
            const int kk = kc + ks * 32;
            if (kk + 32 < S_) __builtin_prefetch(&Arow[kk + 32], 0, 1);

            v16bf af;
#pragma unroll
            for (int j = 0; j < 8; ++j) {
                af[j]     = (bf16)Arow[kk + akb + j];
                af[8 + j] = (bf16)Arow[kk + akb + 8 + j];
            }
#pragma unroll
            for (int t = 0; t < 4; ++t) {
                v16bf bfr;
#pragma unroll
                for (int j = 0; j < 16; ++j) bfr[j] = ldsV[cur][ks * 32 + bkb + j][t * 16 + col];
                acc[t] = __builtin_amdgcn_wmma_f32_16x16x32_bf16(false, af, false, bfr,
                                                                 (short)0, acc[t], false, false);
            }
        }
        __syncthreads();
#if HAS_TDM
        if (wave == 0 && more) __builtin_amdgcn_s_wait_tensorcnt(0);
#endif
        __syncthreads();
    }

    const int mrow = mtile + ((lane >> 4) << 3);
#pragma unroll
    for (int t = 0; t < 4; ++t)
#pragma unroll
        for (int e = 0; e < 8; ++e)
            AOut[((long)b * S_ + mrow + e) * D_ + h * DK_ + t * 16 + col] = (bf16)acc[t][e];
}

// ---------------------------------------------------------------------------
extern "C" void kernel_launch(void* const* d_in, const int* in_sizes, int n_in,
                              void* d_out, int out_size, void* d_ws, size_t ws_size,
                              hipStream_t stream)
{
    (void)in_sizes; (void)n_in; (void)out_size; (void)ws_size;

    // Reference quirk: q projection is applied to the k input; q (d_in[0]) unused.
    const float* k_in = (const float*)d_in[1];
    const float* v_in = (const float*)d_in[2];
    const float* w_q  = (const float*)d_in[3];
    const float* b_q  = (const float*)d_in[4];
    const float* w_k  = (const float*)d_in[5];
    const float* b_k  = (const float*)d_in[6];
    const float* w_v  = (const float*)d_in[7];
    const float* b_v  = (const float*)d_in[8];
    const float* w_o  = (const float*)d_in[9];
    const float* b_o  = (const float*)d_in[10];

    float* out  = (float*)d_out;                   // [B,S,D]
    float* attn = out + (long)B_ * S_ * D_;        // [B,H,S,S]

    const long ND = (long)B_ * S_ * D_;            // 4,194,304
    const long NW = (long)D_ * D_;                 // 1,048,576
    char* ws = (char*)d_ws;
    bf16* kbf = (bf16*)ws; ws += ND * 2;
    bf16* vbf = (bf16*)ws; ws += ND * 2;
    bf16* wqb = (bf16*)ws; ws += NW * 2;
    bf16* wkb = (bf16*)ws; ws += NW * 2;
    bf16* wvb = (bf16*)ws; ws += NW * 2;
    bf16* wob = (bf16*)ws; ws += NW * 2;
    bf16* Qh  = (bf16*)ws; ws += ND * 2;
    bf16* Kh  = (bf16*)ws; ws += ND * 2;
    bf16* Vh  = (bf16*)ws; ws += ND * 2;
    bf16* AO  = (bf16*)ws; ws += ND * 2;           // ~56 MB total

    const dim3 blk(256);

    // 1) downconvert activations + weights
    cvt_f32_bf16<<<(int)((ND + 255) / 256), blk, 0, stream>>>(k_in, kbf, (int)ND);
    cvt_f32_bf16<<<(int)((ND + 255) / 256), blk, 0, stream>>>(v_in, vbf, (int)ND);
    cvt_f32_bf16<<<(int)((NW + 255) / 256), blk, 0, stream>>>(w_q, wqb, (int)NW);
    cvt_f32_bf16<<<(int)((NW + 255) / 256), blk, 0, stream>>>(w_k, wkb, (int)NW);
    cvt_f32_bf16<<<(int)((NW + 255) / 256), blk, 0, stream>>>(w_v, wvb, (int)NW);
    cvt_f32_bf16<<<(int)((NW + 255) / 256), blk, 0, stream>>>(w_o, wob, (int)NW);

    // 2) projections: [4096,1024] = X @ W^T + b   (Q and K both from k input)
    const dim3 gP(D_ / 128, (B_ * S_) / 128, 1);
    gemm_xwT_bf16<<<gP, blk, 0, stream>>>(kbf, D_, wqb, D_, b_q, 1.f, Qh, nullptr, D_,
                                          B_ * S_, D_, D_, 0);
    gemm_xwT_bf16<<<gP, blk, 0, stream>>>(kbf, D_, wkb, D_, b_k, 1.f, Kh, nullptr, D_,
                                          B_ * S_, D_, D_, 0);
    gemm_xwT_bf16<<<gP, blk, 0, stream>>>(vbf, D_, wvb, D_, b_v, 1.f, Vh, nullptr, D_,
                                          B_ * S_, D_, D_, 0);

    // 3) scores = (Qh Kh^T) / 8 per (b,h), f32 straight into attn output
    const dim3 gS(S_ / 128, S_ / 128, B_ * H_);
    gemm_xwT_bf16<<<gS, blk, 0, stream>>>(Qh, D_, Kh, D_, nullptr, 0.125f,
                                          nullptr, attn, S_, S_, S_, DK_, 1);

    // 4) softmax rows in place
    softmax_rows<<<B_ * H_ * S_, blk, 0, stream>>>(attn);

    // 5) attn @ V per (b,h)
    attn_pv<<<dim3(S_ / 128, B_ * H_), blk, 0, stream>>>(attn, Vh, AO);

    // 6) output projection, f32 out + bias
    gemm_xwT_bf16<<<gP, blk, 0, stream>>>(AO, D_, wob, D_, b_o, 1.f, nullptr, out, D_,
                                          B_ * S_, D_, D_, 0);
}
